// SOIA_DOD_91250875170998
// MI455X (gfx1250) — matrix operand order repaired and checked by reading, not compiled, not run on hardware
//
#include <hip/hip_runtime.h>
#include <hip/hip_bf16.h>
#include <cstdint>
#include <cmath>

// ---------------------------------------------------------------------------
// MI455X (gfx1250): all GEMMs via v_wmma_f32_16x16x32_bf16, wave32.
// Main GEMM: each wave computes a 16x64 output strip (4 accumulators),
// reusing one A fragment across 4 B tiles -> 4 WMMA per 2 b128 A loads + 64 16-bit B loads.
// All inner-loop loads are branchless (A: 128-bit vector, B: 16-bit scalar); workspace
// padding guarantees in-bounds accesses (garbage only feeds masked outputs).
// ---------------------------------------------------------------------------

typedef unsigned short u16b;
typedef __attribute__((ext_vector_type(16))) __bf16 v16bf;
typedef __attribute__((ext_vector_type(8)))  float  v8f;

union Frag { u16b u[16]; uint4 q[2]; v16bf v; };
static_assert(sizeof(Frag) == 32, "frag size");

#define DEVINL __device__ __forceinline__

DEVINL u16b f2bf(float f) {
  unsigned int u = __float_as_uint(f);
  u += 0x7fffu + ((u >> 16) & 1u);      // round-to-nearest-even
  return (u16b)(u >> 16);
}

DEVINL v8f wmma_bf16(const Frag& a, const Frag& b, v8f acc) {
  return __builtin_amdgcn_wmma_f32_16x16x32_bf16(
      false, a.v, false, b.v, (short)0, acc, false, false);
}

// A fragment (16x32 bf16, ISA 7.12.2): two contiguous 8-element runs per lane.
DEVINL void load_a(Frag& f, const u16b* rowp, int half) {
  const u16b* p = rowp + half * 8;
  f.q[0] = *(const uint4*)(p);
  f.q[1] = *(const uint4*)(p + 16);
}
// B fragment (32x16 bf16): lane holds column l16, rows 8*half+{0..7,16..23} (p pre-offset).
DEVINL void load_b(Frag& f, const u16b* p, size_t ld) {
  #pragma unroll
  for (int s = 0; s < 16; ++s) f.u[s] = p[(size_t)((s & 7) + (s >> 3) * 16) * ld];
}

// ---------------------------------------------------------------------------
// Generic GEMM: C[M,N] = act(A * B + bias) (+resid). Wave = 16x64 strip.
// Host-side guarantees: M % 16 == 0, K % 32 == 0, ldb % 64 == 0 with zero
// padding in B columns [N, ldb). A-row remap for padded-row operands:
//   arow = (m / a_rpb) * a_bs + m % a_rpb    (compact: a_rpb=M, a_bs=0)
//   crow = (m / c_rpb) * c_bs + c_r0 + m % c_rpb
// ---------------------------------------------------------------------------
__global__ void gemm_bf16_wmma(
    const u16b* __restrict__ A, int lda,
    const u16b* __restrict__ Bm, int ldb,
    const float* __restrict__ bias,
    const float* __restrict__ resid, int ldr,
    float* __restrict__ Cf, u16b* __restrict__ Cbf, int ldc,
    int M, int N, int K, int act,
    int a_rpb, int a_bs, int c_rpb, int c_bs, int c_r0)
{
  int lane = threadIdx.x;
  int n0 = (blockIdx.x * blockDim.y + threadIdx.y) * 64;
  int m0 = blockIdx.y * 16;
  if (n0 >= ldb || m0 >= M) return;             // wave-uniform
  int l16 = lane & 15, half = lane >> 4;

  int am = m0 + l16;
  size_t arow = (size_t)(am / a_rpb) * a_bs + (am % a_rpb);
  const u16b* ap = A + arow * lda;
  const u16b* bp = Bm + (size_t)(half * 8) * ldb + n0 + l16;

  v8f acc[4] = {};
  for (int k0 = 0; k0 < K; k0 += 32) {
    Frag a;
    load_a(a, ap + k0, half);
    const u16b* bk = bp + (size_t)k0 * ldb;
    Frag b[4];
    #pragma unroll
    for (int t = 0; t < 4; ++t) load_b(b[t], bk + t * 16, (size_t)ldb);
    #pragma unroll
    for (int t = 0; t < 4; ++t) acc[t] = wmma_bf16(a, b[t], acc[t]);
  }

  #pragma unroll
  for (int t = 0; t < 4; ++t) {
    int n = n0 + t * 16 + l16;
    if (n < N) {
      float bv = bias ? bias[n] : 0.0f;
      #pragma unroll
      for (int j = 0; j < 8; ++j) {
        int m = m0 + half * 8 + j;
        float v = acc[t][j] + bv;
        if (act == 1)      v = fmaxf(v, 0.0f);                              // relu
        else if (act == 2) v = v / (1.0f + __expf(-1.702f * v));            // quick_gelu
        else if (act == 3) v = fmaxf(v, 0.0f) + log1pf(__expf(-fabsf(v)));  // softplus
        if (resid) v += resid[(size_t)m * ldr + n];
        size_t crow = (size_t)(m / c_rpb) * c_bs + c_r0 + (m % c_rpb);
        if (Cf)  Cf[crow * ldc + n]  = v;
        if (Cbf) Cbf[crow * ldc + n] = f2bf(v);
      }
    }
  }
}

// scores[z, i, j] = sum_d Q[b,i,h*hd+d] * K[b,j,h*hd+d]; padded stride Sp.
__global__ void attn_qk_wmma(const u16b* __restrict__ qkv,
                             float* __restrict__ scores,
                             int S, int Sp, int Dm, int H, int hd)
{
  int lane = threadIdx.x;
  int j0 = (blockIdx.x * blockDim.y + threadIdx.y) * 16;
  int i0 = blockIdx.y * 16;
  int z  = blockIdx.z;
  int b = z / H, h = z % H;
  int l16 = lane & 15, half = lane >> 4;
  int tDm = 3 * Dm;
  const u16b* Qp = qkv + ((size_t)b * S + i0 + l16) * tDm + h * hd;
  const u16b* Kp = qkv + (size_t)b * S * tDm + Dm + h * hd;

  v8f acc = {};
  for (int k0 = 0; k0 < hd; k0 += 32) {
    Frag a, bb;
    load_a(a, Qp + k0, half);
    int kk = k0 + half * 8;
    #pragma unroll
    for (int e = 0; e < 16; ++e)
      bb.u[e] = Kp[(size_t)(j0 + l16) * tDm + kk + (e & 7) + (e >> 3) * 16];
    acc = wmma_bf16(a, bb, acc);
  }
  int j = j0 + l16;
  #pragma unroll
  for (int r = 0; r < 8; ++r) {
    int i = i0 + half * 8 + r;
    scores[((size_t)z * Sp + i) * Sp + j] = acc[r];
  }
}

// ctx[b, i, h*hd+n] = sum_j P[z,i,j] * V[b,j,h*hd+n]; ctx rows padded to Sq16.
__global__ void attn_av_wmma(const u16b* __restrict__ probs,
                             const u16b* __restrict__ qkv,
                             u16b* __restrict__ ctx,
                             int S, int Sp, int Sq16, int Dm, int H, int hd)
{
  int lane = threadIdx.x;
  int n0 = (blockIdx.x * blockDim.y + threadIdx.y) * 16;
  int i0 = blockIdx.y * 16;
  int z  = blockIdx.z;
  int b = z / H, h = z % H;
  int l16 = lane & 15, half = lane >> 4;
  int tDm = 3 * Dm;
  const u16b* Pp = probs + ((size_t)z * Sp + i0 + l16) * Sp;
  const u16b* Vp = qkv + ((size_t)b * S + half * 8) * tDm + 2 * Dm + h * hd + n0 + l16;

  v8f acc = {};
  for (int k0 = 0; k0 < S; k0 += 32) {      // probs zero-padded to Sp in K
    Frag a, bb;
    load_a(a, Pp + k0, half);
    load_b(bb, Vp + (size_t)k0 * tDm, (size_t)tDm);
    acc = wmma_bf16(a, bb, acc);
  }
  int n = n0 + l16;
  #pragma unroll
  for (int r = 0; r < 8; ++r) {
    int i = i0 + half * 8 + r;
    ctx[((size_t)b * Sq16 + i) * Dm + h * hd + n] = f2bf(acc[r]);
  }
}

// wave-per-row layernorm (Dm <= 768); writes f32 and/or bf16
__global__ void layernorm_rows(const float* __restrict__ x,
                               const float* __restrict__ g, const float* __restrict__ bta,
                               float* __restrict__ yf, u16b* __restrict__ ybf,
                               int rows, int Dm)
{
  int row = blockIdx.x * blockDim.y + threadIdx.y;
  if (row >= rows) return;
  int lane = threadIdx.x;
  const float* xr = x + (size_t)row * Dm;
  float v[24];
  int cnt = 0; float s = 0.0f;
  for (int j = lane; j < Dm; j += 32) { float t = xr[j]; v[cnt++] = t; s += t; }
  for (int o = 16; o > 0; o >>= 1) s += __shfl_xor(s, o, 32);
  float mean = s / (float)Dm;
  float s2 = 0.0f;
  for (int c = 0; c < cnt; ++c) { float d = v[c] - mean; s2 += d * d; }
  for (int o = 16; o > 0; o >>= 1) s2 += __shfl_xor(s2, o, 32);
  float inv = rsqrtf(s2 / (float)Dm + 1e-5f);
  cnt = 0;
  for (int j = lane; j < Dm; j += 32) {
    float t = (v[cnt++] - mean) * inv * g[j] + bta[j];
    if (yf)  yf[(size_t)row * Dm + j]  = t;
    if (ybf) ybf[(size_t)row * Dm + j] = f2bf(t);
  }
}

// softmax over rows (z,i): scores [BH, Sp, Sp] f32 -> probs bf16, zero-padding
// columns [S, Sp). S <= 256.
__global__ void softmax_rows(const float* __restrict__ x, u16b* __restrict__ y,
                             int BHS, int S, int Sp, float scale)
{
  int r = blockIdx.x * blockDim.y + threadIdx.y;
  if (r >= BHS) return;
  int z = r / S, i = r % S;
  int lane = threadIdx.x;
  const float* xr = x + ((size_t)z * Sp + i) * Sp;
  u16b* yr = y + ((size_t)z * Sp + i) * Sp;
  float v[8];
  int cnt = 0; float mx = -1e30f;
  for (int j = lane; j < S; j += 32) { float t = xr[j] * scale; v[cnt++] = t; mx = fmaxf(mx, t); }
  for (int o = 16; o > 0; o >>= 1) mx = fmaxf(mx, __shfl_xor(mx, o, 32));
  float sum = 0.0f;
  for (int c = 0; c < cnt; ++c) { float e = __expf(v[c] - mx); v[c] = e; sum += e; }
  for (int o = 16; o > 0; o >>= 1) sum += __shfl_xor(sum, o, 32);
  float invs = 1.0f / sum;
  cnt = 0;
  for (int j = lane; j < Sp; j += 32)
    yr[j] = (j < S) ? f2bf(v[cnt++] * invs) : (u16b)0;
}

// w[N,K] f32 -> wt[Kp,Np] bf16, zero-padded
__global__ void transpose_cast(const float* __restrict__ w, u16b* __restrict__ wt,
                               int N, int K, int Np, int Kp)
{
  int idx = blockIdx.x * blockDim.x + threadIdx.x;
  if (idx >= Np * Kp) return;
  int k = idx / Np, n = idx % Np;
  wt[idx] = (k < K && n < N) ? f2bf(w[(size_t)n * K + k]) : (u16b)0;
}

// dst[r, c<ldd] bf16 = (c<ncols) ? src[(b*bs + r0 + rr), c] : 0
__global__ void cast_gather_rows(const float* __restrict__ src, u16b* __restrict__ dst,
                                 int nrows, int ncols, int ldd, int rpb, int bs, int r0)
{
  int idx = blockIdx.x * blockDim.x + threadIdx.x;
  if (idx >= nrows * ldd) return;
  int r = idx / ldd, c = idx % ldd;
  int b = r / rpb, rr = r % rpb;
  dst[idx] = (c < ncols) ? f2bf(src[((size_t)b * bs + r0 + rr) * ncols + c]) : (u16b)0;
}

__global__ void patch_extract(const float* __restrict__ img, u16b* __restrict__ patches, int B)
{
  int idx = blockIdx.x * blockDim.x + threadIdx.x;
  int total = B * 196 * 768;
  if (idx >= total) return;
  int row = idx / 768, col = idx % 768;
  int b = row / 196, g = row % 196;
  int gy = g / 14, gx = g % 14;
  int c = col / 256, rem = col % 256;
  int py = rem / 16, px = rem % 16;
  patches[idx] = f2bf(img[(((size_t)b * 3 + c) * 224 + (gy * 16 + py)) * 224 + (gx * 16 + px)]);
}

__global__ void build_x0(const float* __restrict__ pe, const float* __restrict__ cls_emb,
                         const float* __restrict__ pos, float* __restrict__ x, int B)
{
  int idx = blockIdx.x * blockDim.x + threadIdx.x;
  int total = B * 197 * 768;
  if (idx >= total) return;
  int row = idx / 768, col = idx % 768;
  int b = row / 197, s = row % 197;
  float v = (s == 0) ? cls_emb[col] : pe[((size_t)b * 196 + (s - 1)) * 768 + col];
  x[idx] = v + pos[(size_t)s * 768 + col];
}

__global__ void gather_cls(const float* __restrict__ table, const int* __restrict__ cls,
                           float* __restrict__ out, int R, int D)
{
  int idx = blockIdx.x * blockDim.x + threadIdx.x;
  if (idx >= R * D) return;
  int r = idx / D, c = idx % D;
  out[idx] = table[(size_t)cls[r] * D + c];
}

// ---------------------------------------------------------------------------
static inline int cdiv(int a, int b) { return (a + b - 1) / b; }
static inline int rup(int a, int b) { return cdiv(a, b) * b; }

extern "C" void kernel_launch(void* const* d_in, const int* in_sizes, int n_in,
                              void* d_out, int out_size, void* d_ws, size_t ws_size,
                              hipStream_t stream) {
  (void)in_sizes; (void)n_in; (void)out_size; (void)ws_size;
  const int Bn = 16, Wd = 768, Ll = 11, Hc = 12, Gg = 196, Sc = 197;
  const int Dd = 512, ELn = 6, EHn = 8, FFe = 2048, MDn = 50, VCn = 74;

  const float* in_img     = (const float*)d_in[0];
  const float* in_box     = (const float*)d_in[1];
  const int*   in_cls     = (const int*)  d_in[2];
  const float* in_conv1   = (const float*)d_in[3];
  const float* in_clsemb  = (const float*)d_in[4];
  const float* in_posemb  = (const float*)d_in[5];
  const float* in_lnpre_g = (const float*)d_in[6];
  const float* in_lnpre_b = (const float*)d_in[7];
  const float* c_qkv_w = (const float*)d_in[8];   const float* c_qkv_b = (const float*)d_in[9];
  const float* c_out_w = (const float*)d_in[10];  const float* c_out_b = (const float*)d_in[11];
  const float* c_ln1_g = (const float*)d_in[12];  const float* c_ln1_b = (const float*)d_in[13];
  const float* c_ln2_g = (const float*)d_in[14];  const float* c_ln2_b = (const float*)d_in[15];
  const float* c_fc_w  = (const float*)d_in[16];  const float* c_fc_b  = (const float*)d_in[17];
  const float* c_pj_w  = (const float*)d_in[18];  const float* c_pj_b  = (const float*)d_in[19];
  const float* fp_w    = (const float*)d_in[20];  const float* fp_b    = (const float*)d_in[21];
  const float* cls_tab = (const float*)d_in[22];
  const float* bb_w1 = (const float*)d_in[23];    const float* bb_b1 = (const float*)d_in[24];
  const float* bb_w2 = (const float*)d_in[25];    const float* bb_b2 = (const float*)d_in[26];
  const float* e_qkv_w = (const float*)d_in[27];  const float* e_qkv_b = (const float*)d_in[28];
  const float* e_out_w = (const float*)d_in[29];  const float* e_out_b = (const float*)d_in[30];
  const float* e_ln1_g = (const float*)d_in[31];  const float* e_ln1_b = (const float*)d_in[32];
  const float* e_ln2_g = (const float*)d_in[33];  const float* e_ln2_b = (const float*)d_in[34];
  const float* e_ff1_w = (const float*)d_in[35];  const float* e_ff1_b = (const float*)d_in[36];
  const float* e_ff2_w = (const float*)d_in[37];  const float* e_ff2_b = (const float*)d_in[38];
  const float* encf_g = (const float*)d_in[39];   const float* encf_b = (const float*)d_in[40];
  const float* obj_w  = (const float*)d_in[41];   const float* obj_b  = (const float*)d_in[42];
  const float* verb_w = (const float*)d_in[43];   const float* verb_b = (const float*)d_in[44];
  const float* ttc_w1 = (const float*)d_in[45];   const float* ttc_b1 = (const float*)d_in[46];
  const float* ttc_w2 = (const float*)d_in[47];   const float* ttc_b2 = (const float*)d_in[48];
  const float* ttc_w3 = (const float*)d_in[49];   const float* ttc_b3 = (const float*)d_in[50];
  float* out = (float*)d_out;

  // ---- workspace carve-up (padded) ----
  const size_t ACT_MAX = (size_t)Bn * Sc * Wd;                 // residual streams
  const size_t QKV_MAX = (size_t)(Bn * Sc + 32) * 3 * Wd;      // +32 pad rows
  const size_t SCO_MAX = (size_t)Bn * Hc * 224 * 224;          // >= enc 16*8*256*256
  const size_t CTX_MAX = (size_t)Bn * 256 * Wd;                // padded rows per batch
  const size_t FF_MAX  = (size_t)Bn * Sc * 4 * Wd;
  const size_t W_MAX   = (size_t)(4 * Wd) * Wd;
  const size_t SM_SZ   = (size_t)Bn * MDn * Dd;

  char* ws = (char*)d_ws; size_t off = 0;
  auto af32 = [&](size_t n) { float* p = (float*)(ws + off); off = (off + n * 4 + 255) & ~(size_t)255; return p; };
  auto au16 = [&](size_t n) { u16b* p = (u16b*)(ws + off); off = (off + n * 2 + 255) & ~(size_t)255; return p; };

  float* X      = af32(ACT_MAX);
  float* X2     = af32(ACT_MAX);
  float* SCORES = af32(SCO_MAX);       // aliased: PEOUT / CLSRES
  u16b* HBF     = au16(ACT_MAX);
  u16b* QKVBF   = au16(QKV_MAX);
  u16b* PROBS   = au16(SCO_MAX);
  u16b* CTXBF   = au16(CTX_MAX);
  u16b* FFBF    = au16(FF_MAX);        // aliased: PATCHBF
  u16b* WBF     = au16(W_MAX);
  u16b* SM1     = au16(SM_SZ);
  u16b* SM2     = au16(SM_SZ);
  u16b* SM3     = au16(SM_SZ);
  float* PEOUT  = SCORES;
  float* CLSRES = SCORES;
  u16b* PATCHBF = FFBF;

  auto ew = [&](int n) { return dim3(cdiv(n, 256)); };
  auto launch_tc = [&](const float* w, u16b* wt, int N, int K, int Np, int Kp) {
    transpose_cast<<<ew(Np * Kp), 256, 0, stream>>>(w, wt, N, K, Np, Kp);
  };
  auto launch_gemm = [&](const u16b* A, int lda, const u16b* Bm, int ldb,
                         const float* bias, const float* resid, int ldr,
                         float* Cf, u16b* Cbf, int ldc,
                         int M, int N, int K, int act,
                         int a_rpb, int a_bs, int crpb, int cbs, int cr0) {
    if (a_rpb <= 0) { a_rpb = M; a_bs = 0; }
    if (crpb <= 0)  { crpb = M; cbs = 0; cr0 = 0; }
    dim3 blk(32, 4), grd(cdiv(ldb / 64, 4), M / 16, 1);
    gemm_bf16_wmma<<<grd, blk, 0, stream>>>(A, lda, Bm, ldb, bias, resid, ldr,
                                            Cf, Cbf, ldc, M, N, K, act,
                                            a_rpb, a_bs, crpb, cbs, cr0);
  };
  auto launch_ln = [&](const float* x, const float* g, const float* b,
                       float* yf, u16b* ybf, int rows, int Dm) {
    dim3 blk(32, 8), grd(cdiv(rows, 8));
    layernorm_rows<<<grd, blk, 0, stream>>>(x, g, b, yf, ybf, rows, Dm);
  };

  auto layer = [&](float*& cur, float*& alt, int S, int Dm, int H, int FFd, int mlp_act,
                   const float* qkv_w, const float* qkv_b,
                   const float* ow, const float* ob,
                   const float* l1g, const float* l1b,
                   const float* l2g, const float* l2b,
                   const float* fw, const float* fb,
                   const float* pw, const float* pb) {
    int M = Bn * S, hd = Dm / H, tDm = 3 * Dm;
    int Sp = rup(S, 32), Sq16 = rup(S, 16);
    float scale = 1.0f / sqrtf((float)hd);
    launch_ln(cur, l1g, l1b, nullptr, HBF, M, Dm);
    launch_tc(qkv_w, WBF, tDm, Dm, tDm, Dm);
    launch_gemm(HBF, Dm, WBF, tDm, qkv_b, nullptr, 0, nullptr, QKVBF, tDm,
                M, tDm, Dm, 0, 0, 0, 0, 0, 0);
    {
      dim3 blk(32, 4), grd(cdiv(Sq16 / 16, 4), Sq16 / 16, Bn * H);
      attn_qk_wmma<<<grd, blk, 0, stream>>>(QKVBF, SCORES, S, Sp, Dm, H, hd);
    }
    {
      dim3 blk(32, 8), grd(cdiv(Bn * H * S, 8));
      softmax_rows<<<grd, blk, 0, stream>>>(SCORES, PROBS, Bn * H * S, S, Sp, scale);
    }
    {
      dim3 blk(32, 4), grd(cdiv(hd / 16, 4), Sq16 / 16, Bn * H);
      attn_av_wmma<<<grd, blk, 0, stream>>>(PROBS, QKVBF, CTXBF, S, Sp, Sq16, Dm, H, hd);
    }
    launch_tc(ow, WBF, Dm, Dm, Dm, Dm);
    launch_gemm(CTXBF, Dm, WBF, Dm, ob, cur, Dm, alt, nullptr, Dm,
                M, Dm, Dm, 0, S, Sq16, 0, 0, 0);
    { float* t = cur; cur = alt; alt = t; }
    launch_ln(cur, l2g, l2b, nullptr, HBF, M, Dm);
    launch_tc(fw, WBF, FFd, Dm, FFd, Dm);
    launch_gemm(HBF, Dm, WBF, FFd, fb, nullptr, 0, nullptr, FFBF, FFd,
                M, FFd, Dm, mlp_act, 0, 0, 0, 0, 0);
    launch_tc(pw, WBF, Dm, FFd, Dm, FFd);
    launch_gemm(FFBF, FFd, WBF, Dm, pb, cur, Dm, alt, nullptr, Dm,
                M, Dm, FFd, 0, 0, 0, 0, 0, 0);
    { float* t = cur; cur = alt; alt = t; }
  };

  // ---- CLIP prologue ----
  patch_extract<<<ew(Bn * Gg * Wd), 256, 0, stream>>>(in_img, PATCHBF, Bn);
  launch_tc(in_conv1, WBF, Wd, Wd, Wd, Wd);
  launch_gemm(PATCHBF, Wd, WBF, Wd, nullptr, nullptr, 0, PEOUT, nullptr, Wd,
              Bn * Gg, Wd, Wd, 0, 0, 0, 0, 0, 0);
  build_x0<<<ew(Bn * Sc * Wd), 256, 0, stream>>>(PEOUT, in_clsemb, in_posemb, X, Bn);
  launch_ln(X, in_lnpre_g, in_lnpre_b, X, nullptr, Bn * Sc, Wd);

  float* cur = X; float* alt = X2;
  for (int i = 0; i < Ll; ++i)
    layer(cur, alt, Sc, Wd, Hc, 4 * Wd, /*quick_gelu*/2,
          c_qkv_w + (size_t)i * 3 * Wd * Wd, c_qkv_b + (size_t)i * 3 * Wd,
          c_out_w + (size_t)i * Wd * Wd,     c_out_b + (size_t)i * Wd,
          c_ln1_g + (size_t)i * Wd, c_ln1_b + (size_t)i * Wd,
          c_ln2_g + (size_t)i * Wd, c_ln2_b + (size_t)i * Wd,
          c_fc_w + (size_t)i * 4 * Wd * Wd,  c_fc_b + (size_t)i * 4 * Wd,
          c_pj_w + (size_t)i * Wd * 4 * Wd,  c_pj_b + (size_t)i * Wd);

  // ---- encoder stream xe[B,246,512] ----
  const int Se = Gg + MDn;
  float* xe = alt;
  cast_gather_rows<<<ew(Bn * Gg * Wd), 256, 0, stream>>>(cur, HBF, Bn * Gg, Wd, Wd, Gg, Sc, 1);
  launch_tc(fp_w, WBF, Dd, Wd, Dd, Wd);
  launch_gemm(HBF, Wd, WBF, Dd, fp_b, nullptr, 0, xe, nullptr, Dd,
              Bn * Gg, Dd, Wd, 0, 0, 0, Gg, Se, 0);
  cast_gather_rows<<<ew(Bn * MDn * 32), 256, 0, stream>>>(in_box, SM1, Bn * MDn, 4, 32, Bn * MDn, 0, 0);
  launch_tc(bb_w1, WBF, Dd, 4, Dd, 32);
  launch_gemm(SM1, 32, WBF, Dd, bb_b1, nullptr, 0, nullptr, SM2, Dd,
              Bn * MDn, Dd, 32, /*relu*/1, 0, 0, 0, 0, 0);
  gather_cls<<<ew(Bn * MDn * Dd), 256, 0, stream>>>(cls_tab, in_cls, CLSRES, Bn * MDn, Dd);
  launch_tc(bb_w2, WBF, Dd, Dd, Dd, Dd);
  launch_gemm(SM2, Dd, WBF, Dd, bb_b2, CLSRES, Dd, xe, nullptr, Dd,
              Bn * MDn, Dd, Dd, 0, 0, 0, MDn, Se, Gg);

  // ---- encoder ----
  cur = xe; alt = (xe == X) ? X2 : X;
  for (int i = 0; i < ELn; ++i)
    layer(cur, alt, Se, Dd, EHn, FFe, /*relu*/1,
          e_qkv_w + (size_t)i * 3 * Dd * Dd, e_qkv_b + (size_t)i * 3 * Dd,
          e_out_w + (size_t)i * Dd * Dd,     e_out_b + (size_t)i * Dd,
          e_ln1_g + (size_t)i * Dd, e_ln1_b + (size_t)i * Dd,
          e_ln2_g + (size_t)i * Dd, e_ln2_b + (size_t)i * Dd,
          e_ff1_w + (size_t)i * FFe * Dd,    e_ff1_b + (size_t)i * FFe,
          e_ff2_w + (size_t)i * Dd * FFe,    e_ff2_b + (size_t)i * Dd);

  // ---- final LN, extract y, heads ----
  launch_ln(cur, encf_g, encf_b, alt, nullptr, Bn * Se, Dd);
  cast_gather_rows<<<ew(Bn * MDn * Dd), 256, 0, stream>>>(alt, SM1, Bn * MDn, Dd, Dd, MDn, Se, Gg);

  const int Rm = Bn * MDn;
  launch_tc(obj_w, WBF, 1, Dd, 64, Dd);
  launch_gemm(SM1, Dd, WBF, 64, obj_b, nullptr, 0, out, nullptr, 1,
              Rm, 1, Dd, 0, 0, 0, 0, 0, 0);
  launch_tc(verb_w, WBF, VCn, Dd, 128, Dd);
  launch_gemm(SM1, Dd, WBF, 128, verb_b, nullptr, 0, out + Rm, nullptr, VCn,
              Rm, VCn, Dd, 0, 0, 0, 0, 0, 0);
  launch_tc(ttc_w1, WBF, Dd, Dd, Dd, Dd);
  launch_gemm(SM1, Dd, WBF, Dd, ttc_b1, nullptr, 0, nullptr, SM2, Dd,
              Rm, Dd, Dd, 1, 0, 0, 0, 0, 0);
  launch_tc(ttc_w2, WBF, Dd, Dd, Dd, Dd);
  launch_gemm(SM2, Dd, WBF, Dd, ttc_b2, nullptr, 0, nullptr, SM3, Dd,
              Rm, Dd, Dd, 1, 0, 0, 0, 0, 0);
  launch_tc(ttc_w3, WBF, 1, Dd, 64, Dd);
  launch_gemm(SM3, Dd, WBF, 64, ttc_b3, nullptr, 0, out + Rm + Rm * VCn, nullptr, 1,
              Rm, 1, Dd, /*softplus*/3, 0, 0, 0, 0, 0);
}
